// MoELayer_54932631716287
// MI455X (gfx1250) — compile-verified, matrix operand order checked by reading, not run on hardware
//
#include <hip/hip_runtime.h>
#include <hip/hip_bf16.h>
#include <cstdint>
#include <cstddef>

// ---------------------------------------------------------------------------
// MoE (top-2 of 8 experts), N=8192 tokens, D=768, F=3072.
// Sparse dispatch + bf16 WMMA GEMMs (v_wmma_f32_16x16x32_bf16), fp32 accum.
// Block tile 128x128, double-buffered LDS, 8 waves x (32x64) -> 8 WMMA/step.
// ---------------------------------------------------------------------------

typedef __attribute__((ext_vector_type(16))) __bf16 v16bf;
typedef __attribute__((ext_vector_type(8)))  float  v8f;

#define N_TOK  8192
#define DMODEL 768
#define NEXP   8
#define FFN    3072
#define NK_ASN 16384   // N_TOK * 2 assignments (exact total rows)

// fp32 -> bf16 round-to-nearest-even (bit pattern in ushort)
static __device__ __forceinline__ unsigned short f2bf(float f) {
    union { float f; unsigned u; } v; v.f = f;
    unsigned u = v.u;
    unsigned r = (u + 0x7FFFu + ((u >> 16) & 1u)) >> 16;
    return (unsigned short)r;
}

static __device__ __forceinline__ float gelu_exact(float x) {
    return 0.5f * x * (1.0f + erff(x * 0.70710678118654752440f));
}

// Load one 16x32 bf16 fragment (A-layout per CDNA5 ISA; B staged transposed in
// LDS so the identical contiguous access pattern applies). Tile row stride=32.
// lane<16 : row=lane,    elems 0-7 -> K=0..7,  elems 8-15 -> K=16..23
// lane>=16: row=lane-16, elems 0-7 -> K=8..15, elems 8-15 -> K=24..31
static __device__ __forceinline__ v16bf ld_frag(const unsigned short* tile, int lane) {
    int row = lane & 15;
    int hi  = lane >> 4;
    const unsigned short* q = tile + row * 32 + hi * 8;
    union { uint4 u[2]; v16bf v; } t;
    t.u[0] = *(const uint4*)(q);
    t.u[1] = *(const uint4*)(q + 16);
    return t.v;
}

// ------------------------------- init --------------------------------------
__global__ __launch_bounds__(256) void k_init(float* __restrict__ out, int* __restrict__ cnt) {
    int i = blockIdx.x * 256 + threadIdx.x;
    if (i < N_TOK * DMODEL) out[i] = 0.0f;
    if (i < NEXP) cnt[i] = 0;
}

// -------------------- weight convert + transpose to bf16 -------------------
// z < 8 : W1 expert z  [D x F] -> W1bT [F x D]
// z >= 8: W2 expert z-8 [F x D] -> W2bT [D x F]
__global__ __launch_bounds__(256) void k_cvt_w(const float* __restrict__ W1,
                                               const float* __restrict__ W2,
                                               unsigned short* __restrict__ W1bT,
                                               unsigned short* __restrict__ W2bT) {
    __shared__ float tile[32][33];
    int z = blockIdx.z;
    int R, C; const float* src; unsigned short* dst;
    if (z < NEXP) { R = DMODEL; C = FFN;    src = W1 + (size_t)z * R * C;          dst = W1bT + (size_t)z * R * C; }
    else          { R = FFN;    C = DMODEL; src = W2 + (size_t)(z - NEXP) * R * C; dst = W2bT + (size_t)(z - NEXP) * R * C; }
    int xb = blockIdx.x * 32, yb = blockIdx.y * 32;
    if (xb >= C || yb >= R) return;
    int tx = threadIdx.x, ty = threadIdx.y;     // 32 x 8
#pragma unroll
    for (int i = 0; i < 4; ++i) {
        int r = yb + ty + i * 8;
        tile[ty + i * 8][tx] = src[(size_t)r * C + xb + tx];
    }
    __syncthreads();
#pragma unroll
    for (int i = 0; i < 4; ++i) {
        int c = xb + ty + i * 8;
        dst[(size_t)c * R + yb + tx] = f2bf(tile[tx][ty + i * 8]);
    }
}

// ------------------------------- router ------------------------------------
__global__ __launch_bounds__(256) void k_router(const float* __restrict__ x,
                                                const float* __restrict__ Wg,
                                                const float* __restrict__ bg,
                                                int* __restrict__ topi,
                                                float* __restrict__ topg,
                                                int* __restrict__ cnt) {
    int wid  = threadIdx.x >> 5;
    int lane = threadIdx.x & 31;
    int n = blockIdx.x * 8 + wid;
    float acc[NEXP];
#pragma unroll
    for (int e = 0; e < NEXP; ++e) acc[e] = 0.0f;
    const float* xr = x + (size_t)n * DMODEL;
    for (int d = lane; d < DMODEL; d += 32) {
        float xv = xr[d];
        const float* wr = Wg + (size_t)d * NEXP;
#pragma unroll
        for (int e = 0; e < NEXP; ++e) acc[e] += xv * wr[e];
    }
#pragma unroll
    for (int off = 16; off > 0; off >>= 1)
#pragma unroll
        for (int e = 0; e < NEXP; ++e) acc[e] += __shfl_xor(acc[e], off);
    if (lane == 0) {
        float l[NEXP], g[NEXP];
        float mx = -3.4e38f;
#pragma unroll
        for (int e = 0; e < NEXP; ++e) { l[e] = acc[e] + bg[e]; mx = fmaxf(mx, l[e]); }
        float s = 0.0f;
#pragma unroll
        for (int e = 0; e < NEXP; ++e) { g[e] = __expf(l[e] - mx); s += g[e]; }
        float inv = 1.0f / s;
#pragma unroll
        for (int e = 0; e < NEXP; ++e) g[e] *= inv;
        int i0 = 0;
#pragma unroll
        for (int e = 1; e < NEXP; ++e) if (g[e] > g[i0]) i0 = e;
        int i1 = (i0 == 0) ? 1 : 0;
#pragma unroll
        for (int e = 0; e < NEXP; ++e) if (e != i0 && g[e] > g[i1]) i1 = e;
        topi[n * 2 + 0] = i0;  topg[n * 2 + 0] = g[i0];
        topi[n * 2 + 1] = i1;  topg[n * 2 + 1] = g[i1];
        atomicAdd(&cnt[i0], 1);
        atomicAdd(&cnt[i1], 1);
    }
}

// ------------------------------- scan --------------------------------------
__global__ void k_scan(const int* __restrict__ cnt, int* __restrict__ offs, int* __restrict__ cursor) {
    if (threadIdx.x == 0 && blockIdx.x == 0) {
        int acc = 0;
        for (int e = 0; e < NEXP; ++e) { offs[e] = acc; cursor[e] = acc; acc += cnt[e]; }
        offs[NEXP] = acc;
    }
}

// ------------------------------ dispatch -----------------------------------
__global__ __launch_bounds__(256) void k_dispatch(const int* __restrict__ topi,
                                                  const float* __restrict__ topg,
                                                  int* __restrict__ cursor,
                                                  int* __restrict__ tok_of,
                                                  float* __restrict__ gate_of) {
    int n = blockIdx.x * 256 + threadIdx.x;
    if (n >= N_TOK) return;
#pragma unroll
    for (int k = 0; k < 2; ++k) {
        int e = topi[n * 2 + k];
        int row = atomicAdd(&cursor[e], 1);
        tok_of[row]  = n;
        gate_of[row] = topg[n * 2 + k];
    }
}

// --------------------------- GEMM1: X @ W1^T, GELU -> H --------------------
// Block tile 128(M) x 128(N), K-step 32, double-buffered LDS.
// 8 waves (4x2), each 32x64 -> 2 A-frags x 4 B-frags = 8 WMMA per K-step.
__global__ __launch_bounds__(256) void k_gemm1(const float* __restrict__ x,
                                               const unsigned short* __restrict__ W1bT,
                                               const float* __restrict__ b1,
                                               unsigned short* __restrict__ H,
                                               const int* __restrict__ tok_of,
                                               const int* __restrict__ offs) {
    __shared__ unsigned short Abuf[2][128 * 32];
    __shared__ unsigned short Bbuf[2][128 * 32];

    int e = blockIdx.z;
    int base = offs[e];
    int rows = offs[e + 1] - base;
    int mBase = blockIdx.y * 128;
    if (mBase >= rows) return;
    int fBase = blockIdx.x * 128;

    int tid  = threadIdx.x;
    int lane = tid & 31;
    int wid  = tid >> 5;
    int wm = wid & 3, wn = wid >> 2;

    // loader roles (fixed across K-loop)
    int lr   = tid >> 1;          // 0..127 tile row
    int half = tid & 1;           // which 16-element chunk of the 32-wide K slab
    bool aValid = (mBase + lr) < rows;
    const float* aRow = aValid ? (x + (size_t)tok_of[base + mBase + lr] * DMODEL) : x;
    const unsigned short* bRow =
        W1bT + ((size_t)e * FFN + fBase + lr) * DMODEL;   // row stride = DMODEL

    v8f c[2][4];
#pragma unroll
    for (int mi = 0; mi < 2; ++mi)
#pragma unroll
        for (int ni = 0; ni < 4; ++ni)
#pragma unroll
            for (int v = 0; v < 8; ++v) c[mi][ni][v] = 0.0f;

    // ---- prologue: fill buffer 0 (kk = 0) ----
    {
        unsigned short tmp[16];
        if (aValid) {
            const float4* src = (const float4*)(aRow + half * 16);
#pragma unroll
            for (int q = 0; q < 4; ++q) {
                float4 f = src[q];
                tmp[q * 4 + 0] = f2bf(f.x); tmp[q * 4 + 1] = f2bf(f.y);
                tmp[q * 4 + 2] = f2bf(f.z); tmp[q * 4 + 3] = f2bf(f.w);
            }
        } else {
#pragma unroll
            for (int q = 0; q < 16; ++q) tmp[q] = 0;
        }
        uint4* dA = (uint4*)&Abuf[0][lr * 32 + half * 16];
        dA[0] = ((const uint4*)tmp)[0];
        dA[1] = ((const uint4*)tmp)[1];
        const uint4* sB = (const uint4*)(bRow + half * 16);
        uint4* dB = (uint4*)&Bbuf[0][lr * 32 + half * 16];
        dB[0] = sB[0];
        dB[1] = sB[1];
    }
    __syncthreads();

    for (int kk = 0; kk < DMODEL; kk += 32) {
        int cur = (kk >> 5) & 1;
        bool hasNext = (kk + 32) < DMODEL;

        // 1) fragments from current buffer
        v16bf a[2], b[4];
#pragma unroll
        for (int mi = 0; mi < 2; ++mi)
            a[mi] = ld_frag(&Abuf[cur][(wm * 32 + mi * 16) * 32], lane);
#pragma unroll
        for (int ni = 0; ni < 4; ++ni)
            b[ni] = ld_frag(&Bbuf[cur][(wn * 64 + ni * 16) * 32], lane);

        // 2) issue next tile's global loads (in flight during WMMAs)
        float4 fS[4];
        uint4  bS[2];
        if (hasNext) {
            const float4* srcA = (const float4*)(aRow + kk + 32 + half * 16);
            if (aValid) {
#pragma unroll
                for (int q = 0; q < 4; ++q) fS[q] = srcA[q];
            } else {
#pragma unroll
                for (int q = 0; q < 4; ++q) fS[q] = float4{0.f, 0.f, 0.f, 0.f};
            }
            const uint4* srcB = (const uint4*)(bRow + kk + 32 + half * 16);
            bS[0] = srcB[0];
            bS[1] = srcB[1];
        }

        // 3) 8 WMMAs
#pragma unroll
        for (int mi = 0; mi < 2; ++mi)
#pragma unroll
            for (int ni = 0; ni < 4; ++ni)
                c[mi][ni] = __builtin_amdgcn_wmma_f32_16x16x32_bf16(
                    false, a[mi], false, b[ni], (short)0, c[mi][ni], false, false);

        // 4) stage next tile into the other buffer
        if (hasNext) {
            unsigned short tmp[16];
#pragma unroll
            for (int q = 0; q < 4; ++q) {
                tmp[q * 4 + 0] = f2bf(fS[q].x); tmp[q * 4 + 1] = f2bf(fS[q].y);
                tmp[q * 4 + 2] = f2bf(fS[q].z); tmp[q * 4 + 3] = f2bf(fS[q].w);
            }
            uint4* dA = (uint4*)&Abuf[1 - cur][lr * 32 + half * 16];
            dA[0] = ((const uint4*)tmp)[0];
            dA[1] = ((const uint4*)tmp)[1];
            uint4* dB = (uint4*)&Bbuf[1 - cur][lr * 32 + half * 16];
            dB[0] = bS[0];
            dB[1] = bS[1];
        }
        __syncthreads();
    }

    // epilogue: bias + exact GELU -> H (bf16)
    int hi = lane >> 4, ncl = lane & 15;
#pragma unroll
    for (int mi = 0; mi < 2; ++mi)
#pragma unroll
        for (int ni = 0; ni < 4; ++ni) {
            int col = fBase + wn * 64 + ni * 16 + ncl;
            float bias = b1[(size_t)e * FFN + col];
#pragma unroll
            for (int v = 0; v < 8; ++v) {
                int rloc = wm * 32 + mi * 16 + hi * 8 + v;
                int gr = mBase + rloc;
                if (gr < rows)
                    H[(size_t)(base + gr) * FFN + col] = f2bf(gelu_exact(c[mi][ni][v] + bias));
            }
        }
}

// --------------------------- GEMM2: H @ W2^T -> scatter-add out ------------
__global__ __launch_bounds__(256) void k_gemm2(const unsigned short* __restrict__ H,
                                               const unsigned short* __restrict__ W2bT,
                                               const float* __restrict__ b2,
                                               float* __restrict__ out,
                                               const int* __restrict__ tok_of,
                                               const float* __restrict__ gate_of,
                                               const int* __restrict__ offs) {
    __shared__ unsigned short Abuf[2][128 * 32];
    __shared__ unsigned short Bbuf[2][128 * 32];

    int e = blockIdx.z;
    int base = offs[e];
    int rows = offs[e + 1] - base;
    int mBase = blockIdx.y * 128;
    if (mBase >= rows) return;
    int dBase = blockIdx.x * 128;

    int tid  = threadIdx.x;
    int lane = tid & 31;
    int wid  = tid >> 5;
    int wm = wid & 3, wn = wid >> 2;

    int lr   = tid >> 1;
    int half = tid & 1;
    bool aValid = (mBase + lr) < rows;
    const unsigned short* aRow = H + (size_t)(base + (aValid ? mBase + lr : 0)) * FFN;
    const unsigned short* bRow =
        W2bT + ((size_t)e * DMODEL + dBase + lr) * FFN;   // row stride = FFN

    v8f c[2][4];
#pragma unroll
    for (int mi = 0; mi < 2; ++mi)
#pragma unroll
        for (int ni = 0; ni < 4; ++ni)
#pragma unroll
            for (int v = 0; v < 8; ++v) c[mi][ni][v] = 0.0f;

    // ---- prologue: fill buffer 0 (kk = 0) ----
    {
        uint4* dA = (uint4*)&Abuf[0][lr * 32 + half * 16];
        if (aValid) {
            const uint4* sA = (const uint4*)(aRow + half * 16);
            dA[0] = sA[0];
            dA[1] = sA[1];
        } else {
            uint4 z{0, 0, 0, 0};
            dA[0] = z; dA[1] = z;
        }
        const uint4* sB = (const uint4*)(bRow + half * 16);
        uint4* dB = (uint4*)&Bbuf[0][lr * 32 + half * 16];
        dB[0] = sB[0];
        dB[1] = sB[1];
    }
    __syncthreads();

    for (int kk = 0; kk < FFN; kk += 32) {
        int cur = (kk >> 5) & 1;
        bool hasNext = (kk + 32) < FFN;

        v16bf a[2], b[4];
#pragma unroll
        for (int mi = 0; mi < 2; ++mi)
            a[mi] = ld_frag(&Abuf[cur][(wm * 32 + mi * 16) * 32], lane);
#pragma unroll
        for (int ni = 0; ni < 4; ++ni)
            b[ni] = ld_frag(&Bbuf[cur][(wn * 64 + ni * 16) * 32], lane);

        uint4 aS[2], bS[2];
        if (hasNext) {
            if (aValid) {
                const uint4* sA = (const uint4*)(aRow + kk + 32 + half * 16);
                aS[0] = sA[0];
                aS[1] = sA[1];
            } else {
                uint4 z{0, 0, 0, 0};
                aS[0] = z; aS[1] = z;
            }
            const uint4* sB = (const uint4*)(bRow + kk + 32 + half * 16);
            bS[0] = sB[0];
            bS[1] = sB[1];
        }

#pragma unroll
        for (int mi = 0; mi < 2; ++mi)
#pragma unroll
            for (int ni = 0; ni < 4; ++ni)
                c[mi][ni] = __builtin_amdgcn_wmma_f32_16x16x32_bf16(
                    false, a[mi], false, b[ni], (short)0, c[mi][ni], false, false);

        if (hasNext) {
            uint4* dA = (uint4*)&Abuf[1 - cur][lr * 32 + half * 16];
            dA[0] = aS[0];
            dA[1] = aS[1];
            uint4* dB = (uint4*)&Bbuf[1 - cur][lr * 32 + half * 16];
            dB[0] = bS[0];
            dB[1] = bS[1];
        }
        __syncthreads();
    }

    // epilogue: out[tok] += gate * (y + b2)   (exactly 2 adds/element -> commutative)
    int hi = lane >> 4, ncl = lane & 15;
#pragma unroll
    for (int mi = 0; mi < 2; ++mi)
#pragma unroll
        for (int ni = 0; ni < 4; ++ni) {
            int col = dBase + wn * 64 + ni * 16 + ncl;
            float bias = b2[(size_t)e * DMODEL + col];
#pragma unroll
            for (int v = 0; v < 8; ++v) {
                int rloc = wm * 32 + mi * 16 + hi * 8 + v;
                int gr = mBase + rloc;
                if (gr < rows) {
                    int rid = base + gr;
                    int tok = tok_of[rid];
                    float g = gate_of[rid];
                    atomicAdd(&out[(size_t)tok * DMODEL + col], g * (c[mi][ni][v] + bias));
                }
            }
        }
}

// ------------------------------- launcher ----------------------------------
extern "C" void kernel_launch(void* const* d_in, const int* in_sizes, int n_in,
                              void* d_out, int out_size, void* d_ws, size_t ws_size,
                              hipStream_t stream) {
    const float* x  = (const float*)d_in[0];
    const float* Wg = (const float*)d_in[1];
    const float* bg = (const float*)d_in[2];
    const float* W1 = (const float*)d_in[3];
    const float* b1 = (const float*)d_in[4];
    const float* W2 = (const float*)d_in[5];
    const float* b2 = (const float*)d_in[6];
    float* out = (float*)d_out;

    const size_t SW = (size_t)NEXP * DMODEL * FFN * sizeof(unsigned short); // 37.7 MB
    const size_t HB = (size_t)NK_ASN * FFN * sizeof(unsigned short);        // 100.7 MB

    uint8_t* w = (uint8_t*)d_ws;
    unsigned short* W1bT = (unsigned short*)w;  w += SW;
    unsigned short* W2bT = (unsigned short*)w;  w += SW;
    unsigned short* Hbuf = (unsigned short*)w;  w += HB;
    int*   topi    = (int*)w;    w += (size_t)N_TOK * 2 * sizeof(int);
    float* topg    = (float*)w;  w += (size_t)N_TOK * 2 * sizeof(float);
    int*   tok_of  = (int*)w;    w += (size_t)NK_ASN * sizeof(int);
    float* gate_of = (float*)w;  w += (size_t)NK_ASN * sizeof(float);
    int*   cnt     = (int*)w;    w += 256;
    int*   offs    = (int*)w;    w += 256;
    int*   cursor  = (int*)w;    w += 256;

    // 1. zero output + counters
    k_init<<<(N_TOK * DMODEL + 255) / 256, 256, 0, stream>>>(out, cnt);
    // 2. weights -> bf16, transposed
    k_cvt_w<<<dim3(FFN / 32, FFN / 32, 2 * NEXP), dim3(32, 8), 0, stream>>>(W1, W2, W1bT, W2bT);
    // 3. router (one wave per token)
    k_router<<<N_TOK / 8, 256, 0, stream>>>(x, Wg, bg, topi, topg, cnt);
    // 4. exclusive scan over 8 experts
    k_scan<<<1, 32, 0, stream>>>(cnt, offs, cursor);
    // 5. compact assignments
    k_dispatch<<<(N_TOK + 255) / 256, 256, 0, stream>>>(topi, topg, cursor, tok_of, gate_of);
    // 6. expert up-projection + GELU (grid: 24 x 128 x 8, early-exit on rows)
    k_gemm1<<<dim3(FFN / 128, NK_ASN / 128, NEXP), 256, 0, stream>>>(x, W1bT, b1, Hbuf, tok_of, offs);
    // 7. expert down-projection + weighted scatter (grid: 6 x 128 x 8)
    k_gemm2<<<dim3(DMODEL / 128, NK_ASN / 128, NEXP), 256, 0, stream>>>(Hbuf, W2bT, b2, out, tok_of, gate_of, offs);
}
